// MixSoftmax_13597866459836
// MI455X (gfx1250) — compile-verified, matrix operand order checked by reading, not run on hardware
//
#include <hip/hip_runtime.h>

// Problem dims (fixed by reference)
#define Bdim 4
#define Sdim 1024
#define Hdim 1024
#define Kdim 8
#define Edim 512
#define Cdim 10000
#define M1   (Bdim*Sdim)   // 4096 token rows
#define KE   (Kdim*Edim)   // 4096 latent units
#define RROWS (M1*Kdim)    // 32768 pseudo-rows of decoder GEMM
#define NTILES (Cdim/16)   // 625 column tiles of 16

typedef __attribute__((ext_vector_type(16))) __bf16 v16bf;
typedef __attribute__((ext_vector_type(8)))  __bf16 v8bf;
typedef __attribute__((ext_vector_type(8)))  float  v8f;

static __device__ __forceinline__ unsigned short f2bf(float f) {
  unsigned int u = __builtin_bit_cast(unsigned int, f);
  u += 0x7FFFu + ((u >> 16) & 1u);           // round-to-nearest-even
  return (unsigned short)(u >> 16);
}

// branchless native exp: single v_exp_f32
static __device__ __forceinline__ float fexp(float x) {
  return __builtin_amdgcn_exp2f(x * 1.44269504088896340736f);
}

// branchless tanh: tanh(x) = 1 - 2/(exp(2x)+1); saturates correctly at +-inf
static __device__ __forceinline__ float fast_tanh(float x) {
  float e = fexp(2.0f * x);
  return 1.0f - 2.0f * __builtin_amdgcn_rcpf(e + 1.0f);
}

static __device__ __forceinline__ v16bf cat8(v8bf lo, v8bf hi) {
  return __builtin_shufflevector(lo, hi, 0,1,2,3,4,5,6,7,8,9,10,11,12,13,14,15);
}

static __device__ __forceinline__ v8bf ld8(const unsigned short* p) {
  return *reinterpret_cast<const v8bf*>(p);
}

// CDNA5 async copy: 16 bytes global -> LDS, tracked by ASYNCcnt, no VGPR
// round-trip (ISA 15.18.3 GLOBAL_LOAD_ASYNC_TO_LDS_B128; VDST = LDS byte
// address = low 32 bits of the generic shared pointer).
static __device__ __forceinline__ void async_cp16(void* lds, const void* gptr) {
  unsigned loff = (unsigned)(size_t)lds;
  asm volatile("global_load_async_to_lds_b128 %0, %1, off"
               :: "v"(loff), "v"(gptr) : "memory");
}
static __device__ __forceinline__ void wait_async0() {
  asm volatile("s_wait_asynccnt 0x0" ::: "memory");
}

// ---------------- fp32 -> bf16 conversion (grid-stride) ----------------
__global__ void cvt_bf16(const float* __restrict__ src,
                         unsigned short* __restrict__ dst, int n) {
  int stride = gridDim.x * blockDim.x;
  for (int i = blockIdx.x * blockDim.x + threadIdx.x; i < n; i += stride)
    dst[i] = f2bf(src[i]);
}

// ---------------- priors: softmax over K=8 (tiny, fp32) ----------------
__global__ void priors_kernel(const float* __restrict__ ctx,
                              const float* __restrict__ pw,
                              const float* __restrict__ pb,
                              float* __restrict__ priors) {
  __shared__ float red[Kdim][128];
  int row = blockIdx.x, t = threadIdx.x;
  float acc[Kdim];
#pragma unroll
  for (int k = 0; k < Kdim; ++k) acc[k] = 0.f;
  for (int h = t; h < Hdim; h += 128) {
    float cv = ctx[row * Hdim + h];
#pragma unroll
    for (int k = 0; k < Kdim; ++k) acc[k] += cv * pw[k * Hdim + h];
  }
#pragma unroll
  for (int k = 0; k < Kdim; ++k) red[k][t] = acc[k];
  __syncthreads();
  for (int s = 64; s > 0; s >>= 1) {
    if (t < s)
#pragma unroll
      for (int k = 0; k < Kdim; ++k) red[k][t] += red[k][t + s];
    __syncthreads();
  }
  if (t == 0) {
    float l[Kdim], m = -3.0e38f, z = 0.f;
#pragma unroll
    for (int k = 0; k < Kdim; ++k) { l[k] = red[k][0] + pb[k]; m = fmaxf(m, l[k]); }
#pragma unroll
    for (int k = 0; k < Kdim; ++k) { l[k] = fexp(l[k] - m); z += l[k]; }
#pragma unroll
    for (int k = 0; k < Kdim; ++k) priors[row * Kdim + k] = l[k] / z;
  }
}

// ---------------- latent GEMM: lat16 = bf16(tanh(ctx @ lw^T + b)) ------
// One wave computes one 16x16 tile; A=[4096,1024] bf16, B=[4096,1024] bf16
// (both K-major), both L2-resident (8 MB each).
__global__ void __launch_bounds__(256) latent_gemm(
    const unsigned short* __restrict__ ctx16,
    const unsigned short* __restrict__ lw16,
    const float* __restrict__ lat_b,
    unsigned short* __restrict__ lat16) {
  int lane = threadIdx.x & 31, wave = threadIdx.x >> 5;
  int gid = blockIdx.x * 8 + wave;
  int m0 = (gid >> 8) * 16;          // 256 M tiles
  int n0 = (gid & 255) * 16;         // 256 N tiles
  int lr = lane & 15, hi = lane >> 4;
  const unsigned short* arow = ctx16 + (size_t)(m0 + lr) * Hdim + hi * 8;
  const unsigned short* brow = lw16  + (size_t)(n0 + lr) * Hdim + hi * 8;
  v8f acc = {};
#pragma unroll 4
  for (int kt = 0; kt < Hdim / 32; ++kt) {
    const unsigned short* ap = arow + kt * 32;
    const unsigned short* bp = brow + kt * 32;
    v16bf a = cat8(ld8(ap), ld8(ap + 16));
    v16bf b = cat8(ld8(bp), ld8(bp + 16));
    acc = __builtin_amdgcn_wmma_f32_16x16x32_bf16(false, a, false, b,
                                                  (short)0, acc, false, false);
  }
  int col = n0 + lr;
  float bias = lat_b[col];
#pragma unroll
  for (int j = 0; j < 8; ++j) {
    int row = m0 + j + 8 * hi;       // C-frag: VGPR j, half hi -> row j+8*hi
    lat16[(size_t)row * KE + col] = f2bf(fast_tanh(acc[j] + bias));
  }
}

// ---------------- decoder pass 1: per-row max & sum(exp) ---------------
// Block = 8 waves = 128 pseudo-rows. A tile (16x512) lives in registers
// per wave (16 v16bf frags); dec_w N-tile (16x512 bf16, 16 KB) streamed
// into double-buffered LDS with GLOBAL_LOAD_ASYNC_TO_LDS_B128, so the
// transfer of tile nt+1 overlaps the 16 WMMAs of tile nt.
__global__ void __launch_bounds__(256) dec_pass1(
    const unsigned short* __restrict__ lat16,
    const unsigned short* __restrict__ dec16,
    const float* __restrict__ dec_b,
    float* __restrict__ mrow, float* __restrict__ zrow) {
  __shared__ __align__(16) unsigned short Bs[2][16 * Edim];   // 32 KB
  int t = threadIdx.x;
  int lane = t & 31, wave = t >> 5;
  int lr = lane & 15, hi = lane >> 4;
  int rw = blockIdx.x * 128 + wave * 16;

  const unsigned short* arow = lat16 + (size_t)(rw + lr) * Edim + hi * 8;
  v16bf afrag[16];
#pragma unroll
  for (int kt = 0; kt < 16; ++kt)
    afrag[kt] = cat8(ld8(arow + kt * 32), ld8(arow + kt * 32 + 16));

  float m[8], ssum[8];
#pragma unroll
  for (int j = 0; j < 8; ++j) { m[j] = -3.0e38f; ssum[j] = 0.f; }

  auto issue_tile = [&](int nt, int buf) {
    int n0 = nt * 16;
#pragma unroll
    for (int i = 0; i < 4; ++i) {                 // 16 KB / 256 threads
      int idx = t + i * 256;                      // chunk of 8 ushorts
      int row = idx >> 6, c8 = idx & 63;
      async_cp16(&Bs[buf][row * Edim + c8 * 8],
                 &dec16[(size_t)(nt * 16 + row) * Edim + c8 * 8]);
    }
    (void)n0;
  };

  issue_tile(0, 0);
  for (int nt = 0; nt < NTILES; ++nt) {
    int buf = nt & 1;
    wait_async0();                // my async copies into Bs[buf] landed
    __syncthreads();              // everyone's landed; prior reads of buf^1 done
    if (nt + 1 < NTILES) issue_tile(nt + 1, buf ^ 1);

    v8f acc = {};
#pragma unroll
    for (int kt = 0; kt < 16; ++kt) {
      const unsigned short* bp = &Bs[buf][lr * Edim + kt * 32 + hi * 8];
      v16bf b = cat8(*(const v8bf*)bp, *(const v8bf*)(bp + 16));
      acc = __builtin_amdgcn_wmma_f32_16x16x32_bf16(false, afrag[kt], false, b,
                                                    (short)0, acc, false, false);
    }
    float dbias = dec_b[nt * 16 + lr];
#pragma unroll
    for (int j = 0; j < 8; ++j) {                 // online (max, sumexp)
      float v = acc[j] + dbias;
      float nm = fmaxf(m[j], v);
      ssum[j] = ssum[j] * fexp(m[j] - nm) + fexp(v - nm);
      m[j] = nm;
    }
  }
  // reduce (m,s) across the 16 lanes of each half-wave (row lives in one half)
#pragma unroll
  for (int j = 0; j < 8; ++j) {
    float mj = m[j], sj = ssum[j];
#pragma unroll
    for (int off = 1; off < 16; off <<= 1) {
      float om = __shfl_xor(mj, off, 32);
      float os = __shfl_xor(sj, off, 32);
      float nm = fmaxf(mj, om);
      sj = sj * fexp(mj - nm) + os * fexp(om - nm);
      mj = nm;
    }
    if (lr == 0) {
      int row = rw + j + 8 * hi;
      mrow[row] = mj;
      zrow[row] = sj;
    }
  }
}

// ---------------- decoder pass 2: out = sum_k prior/Z * exp(logit-m) ---
// Mixture sum over K=8 == sum over the 8 accumulator VGPRs of one lane.
__global__ void __launch_bounds__(256) dec_pass2(
    const unsigned short* __restrict__ lat16,
    const unsigned short* __restrict__ dec16,
    const float* __restrict__ dec_b,
    const float* __restrict__ priors,
    const float* __restrict__ mrow, const float* __restrict__ zrow,
    float* __restrict__ out) {
  __shared__ __align__(16) unsigned short Bs[2][16 * Edim];   // 32 KB
  int t = threadIdx.x;
  int lane = t & 31, wave = t >> 5;
  int lr = lane & 15, hi = lane >> 4;
  int rw = blockIdx.x * 128 + wave * 16;

  const unsigned short* arow = lat16 + (size_t)(rw + lr) * Edim + hi * 8;
  v16bf afrag[16];
#pragma unroll
  for (int kt = 0; kt < 16; ++kt)
    afrag[kt] = cat8(ld8(arow + kt * 32), ld8(arow + kt * 32 + 16));

  float mm[8], cf[8];
#pragma unroll
  for (int j = 0; j < 8; ++j) {
    int row = rw + j + 8 * hi;       // row = bs*8 + k, priors flat-indexed
    mm[j] = mrow[row];
    cf[j] = priors[row] * __builtin_amdgcn_rcpf(zrow[row]);
  }
  int bs = (rw >> 3) + hi;           // this half-wave owns one token row

  auto issue_tile = [&](int nt, int buf) {
#pragma unroll
    for (int i = 0; i < 4; ++i) {
      int idx = t + i * 256;
      int row = idx >> 6, c8 = idx & 63;
      async_cp16(&Bs[buf][row * Edim + c8 * 8],
                 &dec16[(size_t)(nt * 16 + row) * Edim + c8 * 8]);
    }
  };

  issue_tile(0, 0);
  for (int nt = 0; nt < NTILES; ++nt) {
    int buf = nt & 1;
    wait_async0();
    __syncthreads();
    if (nt + 1 < NTILES) issue_tile(nt + 1, buf ^ 1);

    v8f acc = {};
#pragma unroll
    for (int kt = 0; kt < 16; ++kt) {
      const unsigned short* bp = &Bs[buf][lr * Edim + kt * 32 + hi * 8];
      v16bf b = cat8(*(const v8bf*)bp, *(const v8bf*)(bp + 16));
      acc = __builtin_amdgcn_wmma_f32_16x16x32_bf16(false, afrag[kt], false, b,
                                                    (short)0, acc, false, false);
    }
    float dbias = dec_b[nt * 16 + lr];
    float o = 0.f;
#pragma unroll
    for (int j = 0; j < 8; ++j)
      o += cf[j] * fexp(acc[j] + dbias - mm[j]);
    out[(size_t)bs * Cdim + nt * 16 + lr] = o;
  }
}

// ----------------------------- launcher --------------------------------
extern "C" void kernel_launch(void* const* d_in, const int* in_sizes, int n_in,
                              void* d_out, int out_size, void* d_ws, size_t ws_size,
                              hipStream_t stream) {
  const float* ctx = (const float*)d_in[0];   // [4096,1024]
  const float* pw  = (const float*)d_in[1];   // [8,1024]
  const float* pb  = (const float*)d_in[2];   // [8]
  const float* lw  = (const float*)d_in[3];   // [4096,1024]
  const float* lb  = (const float*)d_in[4];   // [4096]
  const float* dw  = (const float*)d_in[5];   // [10000,512]
  const float* db  = (const float*)d_in[6];   // [10000]
  float* out = (float*)d_out;                 // [4096,10000]

  // workspace carve-up (~61 MB total)
  char* ws = (char*)d_ws;
  size_t off = 0;
  auto carve = [&](size_t bytes) {
    void* p = ws + off;
    off = (off + bytes + 255) & ~(size_t)255;
    return p;
  };
  unsigned short* ctx16 = (unsigned short*)carve((size_t)M1 * Hdim * 2);
  unsigned short* lw16  = (unsigned short*)carve((size_t)KE * Hdim * 2);
  unsigned short* dec16 = (unsigned short*)carve((size_t)Cdim * Edim * 2);
  unsigned short* lat16 = (unsigned short*)carve((size_t)M1 * KE * 2);
  float* priors = (float*)carve((size_t)M1 * Kdim * 4);
  float* mrow   = (float*)carve((size_t)RROWS * 4);
  float* zrow   = (float*)carve((size_t)RROWS * 4);

  cvt_bf16<<<2048, 256, 0, stream>>>(ctx, ctx16, M1 * Hdim);
  cvt_bf16<<<2048, 256, 0, stream>>>(lw,  lw16,  KE * Hdim);
  cvt_bf16<<<2048, 256, 0, stream>>>(dw,  dec16, Cdim * Edim);

  priors_kernel<<<M1, 128, 0, stream>>>(ctx, pw, pb, priors);

  latent_gemm<<<(256 * 256) / 8, 256, 0, stream>>>(ctx16, lw16, lb, lat16);

  dec_pass1<<<RROWS / 128, 256, 0, stream>>>(lat16, dec16, db, mrow, zrow);
  dec_pass2<<<RROWS / 128, 256, 0, stream>>>(lat16, dec16, db, priors,
                                             mrow, zrow, out);
}
